// Attention_55645596287359
// MI455X (gfx1250) — compile-verified, matrix operand order checked by reading, not run on hardware
//
#include <hip/hip_runtime.h>

typedef __bf16 v16bf __attribute__((ext_vector_type(16)));
typedef float  v8f   __attribute__((ext_vector_type(8)));

#define B_   4
#define NH_  4
#define T_   512
#define D_   256
#define N_   8192
#define BH_  (B_*NH_)

static constexpr float SCALE_ = 0.011048543456039806f; // 8192^-0.5

#define BM  64
#define BN  64
#define KS  32          // K-step per stage
#define LDK 40          // bf16 row length, k-contig layout  (80B rows, 16B chunks)
#define LDM 72          // bf16 row length, m/n-contig layout (144B rows, 16B chunks)
#define ATILE 2560      // max(64*40, 32*72) bf16 elems; 5120B (16B aligned)
#define NTHREADS 128

__device__ __forceinline__ __bf16 f2bf(float x) {
    unsigned u = __builtin_bit_cast(unsigned, x);
    unsigned r = u + 0x7FFFu + ((u >> 16) & 1u);   // round-to-nearest-even
    unsigned short h = (unsigned short)(r >> 16);
    return __builtin_bit_cast(__bf16, h);
}

// ---- gfx1250 async global->LDS copy (ASYNCcnt), 16 bytes per lane ----------
__device__ __forceinline__ void async_copy16(const __bf16* lds, const __bf16* g) {
    asm volatile("global_load_async_to_lds_b128 %0, %1, off"
                 :: "v"((unsigned)(size_t)lds), "v"(g)
                 : "memory");
}
__device__ __forceinline__ void wait_async() {
    asm volatile("s_wait_asynccnt 0x0" ::: "memory");
}

// Issue async fills for one 64x32 (or 32x64) operand tile: 256 x 16B chunks.
// KC=true : operand contiguous along K (stride 1); LDS layout [outer][LDK]
// KC=false: operand contiguous along M/N (stride 1); LDS layout [k][LDM]
template <bool KC>
__device__ __forceinline__ void stage_issue(const __bf16* __restrict__ P,
                                            int s_outer, int s_k,
                                            int o0, int k0, __bf16* buf)
{
    const int tid = threadIdx.x;
#pragma unroll
    for (int c = tid; c < 256; c += NTHREADS) {
        if constexpr (KC) {
            int m  = c >> 2;
            int kc = (c & 3) * 8;
            async_copy16(buf + m * LDK + kc,
                         P + (size_t)(o0 + m) * s_outer + (k0 + kc));
        } else {
            int k  = c >> 3;
            int mc = (c & 7) * 8;
            async_copy16(buf + k * LDM + mc,
                         P + (size_t)(k0 + k) * s_k + (o0 + mc));
        }
    }
}

// ---------------------------------------------------------------------------
// Double-buffered async-staged bf16 GEMM block: accumulates
// C[m0:m0+64, n0:n0+64] into acc[4] for the calling wave.
// 128 threads = 4 wave32s, each wave a 32x32 subtile = 2x2 v_wmma_f32_16x16x32_bf16.
// One s_barrier per K-step; tile k+1 async fill overlaps tile k WMMAs.
// ---------------------------------------------------------------------------
template <bool AKC, bool BKC>
__device__ __forceinline__ void gemm_tile(
    const __bf16* __restrict__ A, int sa_m, int sa_k,
    const __bf16* __restrict__ B, int sb_k, int sb_n,
    int m0, int n0, int K,
    __bf16* Ab, __bf16* Bb,                 // each 2*ATILE, 16B aligned
    v8f acc[4])
{
    const int tid  = threadIdx.x;
    const int lane = tid & 31;
    const int wid  = tid >> 5;
    const int wm   = (wid >> 1) * 32;
    const int wn   = (wid & 1) * 32;
    const int ml   = lane & 15;
    const int half = lane >> 4;
    const int nk   = K / KS;

    stage_issue<AKC>(A, sa_m, sa_k, m0, 0, Ab);
    stage_issue<BKC>(B, sb_n, sb_k, n0, 0, Bb);

    for (int it = 0; it < nk; ++it) {
        wait_async();         // own fills for buf[it&1] complete
        __syncthreads();      // all waves' fills visible; prev-iter reads done
        if (it + 1 < nk) {
            stage_issue<AKC>(A, sa_m, sa_k, m0, (it + 1) * KS,
                             Ab + ((it + 1) & 1) * ATILE);
            stage_issue<BKC>(B, sb_n, sb_k, n0, (it + 1) * KS,
                             Bb + ((it + 1) & 1) * ATILE);
        }
        const __bf16* Ac = Ab + (it & 1) * ATILE;
        const __bf16* Bc = Bb + (it & 1) * ATILE;

        // Fragment loads per ISA 16-bit A/B VGPR layout (wave32, 16x16x32)
        v16bf afr[2], bfr[2];
#pragma unroll
        for (int f = 0; f < 2; ++f) {
            const int mr = wm + f * 16 + ml;
            const int nr = wn + f * 16 + ml;
#pragma unroll
            for (int i = 0; i < 8; ++i) {
                int kk = ((i >> 2) << 4) + (half << 3) + ((i & 3) << 1);
                if constexpr (AKC) {
                    afr[f][2*i]   = Ac[mr * LDK + kk];
                    afr[f][2*i+1] = Ac[mr * LDK + kk + 1];
                } else {
                    afr[f][2*i]   = Ac[kk * LDM + mr];
                    afr[f][2*i+1] = Ac[(kk + 1) * LDM + mr];
                }
                if constexpr (BKC) {
                    bfr[f][2*i]   = Bc[nr * LDK + kk];
                    bfr[f][2*i+1] = Bc[nr * LDK + kk + 1];
                } else {
                    bfr[f][2*i]   = Bc[kk * LDM + nr];
                    bfr[f][2*i+1] = Bc[(kk + 1) * LDM + nr];
                }
            }
        }
#pragma unroll
        for (int fm = 0; fm < 2; ++fm)
#pragma unroll
            for (int fn = 0; fn < 2; ++fn)
                acc[fm * 2 + fn] = __builtin_amdgcn_wmma_f32_16x16x32_bf16(
                    false, afr[fm], false, bfr[fn],
                    (short)0, acc[fm * 2 + fn], false, false);
    }
    __syncthreads();   // protect buffers across back-to-back gemm_tile calls
}

// Epilogue index: frag (fm,fn), vgpr r:
//   m = m0+wm+fm*16+(lane>>4)*8+r,  n = n0+wn+fn*16+(lane&15)

// ---------------------------------------------------------------------------
__global__ void __launch_bounds__(256)
rope_kernel(const float* __restrict__ Q, const float* __restrict__ cosv,
            const float* __restrict__ sinv, __bf16* __restrict__ QR)
{
    size_t p = (size_t)blockIdx.x * 256 + threadIdx.x;   // pair index
    const int n2 = N_ / 2;
    int    n    = (int)(p % n2) * 2;
    size_t rest = p / n2;
    int    t    = (int)(rest % T_);
    size_t bh   = rest / T_;
    size_t qi = (bh * T_ + t) * (size_t)N_ + n;
    size_t ci = (size_t)t * N_ + n;
    float qe = Q[qi], qo = Q[qi + 1];
    float ce = cosv[ci], co = cosv[ci + 1];
    float se = sinv[ci], so = sinv[ci + 1];
    QR[qi]     = f2bf(qe * ce - qo * se);
    QR[qi + 1] = f2bf(qo * co + qe * so);
}

__global__ void __launch_bounds__(256)
cvt_kernel(const float* __restrict__ src, __bf16* __restrict__ dst, size_t n)
{
    size_t i = (size_t)blockIdx.x * 256 + threadIdx.x;
    if (i < n) dst[i] = f2bf(src[i]);
}

// ---------------------------------------------------------------------------
// scores = tril(QR . QR^T * SCALE, k=-1), stored bf16. Above-diagonal blocks
// skip the K loop entirely (writes zeros only).
// ---------------------------------------------------------------------------
__global__ void __launch_bounds__(128)
scores_kernel(const __bf16* __restrict__ QR, __bf16* __restrict__ S)
{
    __shared__ __attribute__((aligned(16))) __bf16 Ab[2 * ATILE];
    __shared__ __attribute__((aligned(16))) __bf16 Bb[2 * ATILE];
    const int bh = blockIdx.z;
    const int m0 = blockIdx.y * BM;
    const int n0 = blockIdx.x * BN;
    const __bf16* Qb = QR + (size_t)bh * T_ * N_;
    __bf16* Sb = S + (size_t)bh * T_ * T_;

    v8f acc[4] = {};
    if (n0 <= m0) {  // uniform over block
        // A = QR (MxK, k-contig), B(k,n) = QR[n*N_+k] (k-contig)
        gemm_tile<true, true>(Qb, N_, 1, Qb, 1, N_, m0, n0, N_, Ab, Bb, acc);
    }
    const int tid = threadIdx.x, lane = tid & 31, wid = tid >> 5;
    const int wm = (wid >> 1) * 32, wn = (wid & 1) * 32;
    const int ml = lane & 15, half = lane >> 4;
#pragma unroll
    for (int fm = 0; fm < 2; ++fm)
#pragma unroll
        for (int fn = 0; fn < 2; ++fn)
#pragma unroll
            for (int r = 0; r < 8; ++r) {
                int m = m0 + wm + fm * 16 + half * 8 + r;
                int n = n0 + wn + fn * 16 + ml;
                float v = (n < m) ? acc[fm * 2 + fn][r] * SCALE_ : 0.0f;
                Sb[(size_t)m * T_ + n] = f2bf(v);
            }
}

// ---------------------------------------------------------------------------
// output = scores . V  +  QR . state  (two K-loops fused into one accumulator)
// ---------------------------------------------------------------------------
__global__ void __launch_bounds__(128)
output_kernel(const __bf16* __restrict__ S, const __bf16* __restrict__ Vb,
              const __bf16* __restrict__ QR, const __bf16* __restrict__ STb,
              float* __restrict__ out)
{
    __shared__ __attribute__((aligned(16))) __bf16 Ab[2 * ATILE];
    __shared__ __attribute__((aligned(16))) __bf16 Bb[2 * ATILE];
    const int bh = blockIdx.z;
    const int b  = bh / NH_;
    const int m0 = blockIdx.y * BM;
    const int n0 = blockIdx.x * BN;

    v8f acc[4] = {};
    // scores(TxT, k-contig) . V(TxD, n-contig)
    gemm_tile<true, false>(S + (size_t)bh * T_ * T_, T_, 1,
                           Vb + (size_t)b * T_ * D_, D_, 1,
                           m0, n0, T_, Ab, Bb, acc);
    // QR(TxN, k-contig) . state(NxD, n-contig)
    gemm_tile<true, false>(QR + (size_t)bh * T_ * N_, N_, 1,
                           STb + (size_t)bh * N_ * D_, D_, 1,
                           m0, n0, N_, Ab, Bb, acc);

    float* Ob = out + (size_t)bh * T_ * D_;
    const int tid = threadIdx.x, lane = tid & 31, wid = tid >> 5;
    const int wm = (wid >> 1) * 32, wn = (wid & 1) * 32;
    const int ml = lane & 15, half = lane >> 4;
#pragma unroll
    for (int fm = 0; fm < 2; ++fm)
#pragma unroll
        for (int fn = 0; fn < 2; ++fn)
#pragma unroll
            for (int r = 0; r < 8; ++r) {
                int m = m0 + wm + fm * 16 + half * 8 + r;
                int n = n0 + wn + fn * 16 + ml;
                Ob[(size_t)m * D_ + n] = acc[fm * 2 + fn][r];
            }
}

// ---------------------------------------------------------------------------
// new_state = state + SCALE * QR^T . V   A(m,k)=QR[k*N_+m] (m-contig)
// ---------------------------------------------------------------------------
__global__ void __launch_bounds__(128)
state_kernel(const __bf16* __restrict__ QR, const __bf16* __restrict__ Vb,
             const float* __restrict__ STf, float* __restrict__ out2)
{
    __shared__ __attribute__((aligned(16))) __bf16 Ab[2 * ATILE];
    __shared__ __attribute__((aligned(16))) __bf16 Bb[2 * ATILE];
    const int bh = blockIdx.z;
    const int b  = bh / NH_;
    const int m0 = blockIdx.y * BM;
    const int n0 = blockIdx.x * BN;

    v8f acc[4] = {};
    gemm_tile<false, false>(QR + (size_t)bh * T_ * N_, 1, N_,
                            Vb + (size_t)b * T_ * D_, D_, 1,
                            m0, n0, T_, Ab, Bb, acc);

    const float* Sf = STf + (size_t)bh * N_ * D_;
    float* Ob = out2 + (size_t)bh * N_ * D_;
    const int tid = threadIdx.x, lane = tid & 31, wid = tid >> 5;
    const int wm = (wid >> 1) * 32, wn = (wid & 1) * 32;
    const int ml = lane & 15, half = lane >> 4;
#pragma unroll
    for (int fm = 0; fm < 2; ++fm)
#pragma unroll
        for (int fn = 0; fn < 2; ++fn)
#pragma unroll
            for (int r = 0; r < 8; ++r) {
                int m = m0 + wm + fm * 16 + half * 8 + r;
                int n = n0 + wn + fn * 16 + ml;
                size_t idx = (size_t)m * D_ + n;
                Ob[idx] = Sf[idx] + SCALE_ * acc[fm * 2 + fn][r];
            }
}

// ---------------------------------------------------------------------------
extern "C" void kernel_launch(void* const* d_in, const int* in_sizes, int n_in,
                              void* d_out, int out_size, void* d_ws, size_t ws_size,
                              hipStream_t stream)
{
    (void)in_sizes; (void)n_in; (void)out_size; (void)ws_size;
    const float* Q    = (const float*)d_in[0];
    const float* V    = (const float*)d_in[1];
    const float* ST   = (const float*)d_in[2];
    const float* cosv = (const float*)d_in[3];
    const float* sinv = (const float*)d_in[4];

    float* out  = (float*)d_out;                    // (BH, T, D)
    float* out2 = out + (size_t)BH_ * T_ * D_;      // (BH, N, D)

    // Workspace: QR bf16 | state bf16 | V bf16 | scores bf16  (~211 MB)
    __bf16* qr  = (__bf16*)d_ws;
    __bf16* stb = qr  + (size_t)BH_ * T_ * N_;
    __bf16* vb  = stb + (size_t)BH_ * N_ * D_;
    __bf16* sc  = vb  + (size_t)B_ * T_ * D_;

    const size_t pairs = (size_t)BH_ * T_ * (N_ / 2);
    const size_t st_n  = (size_t)BH_ * N_ * D_;
    const size_t v_n   = (size_t)B_ * T_ * D_;

    rope_kernel<<<dim3((unsigned)(pairs / 256)), 256, 0, stream>>>(Q, cosv, sinv, qr);
    cvt_kernel <<<dim3((unsigned)(st_n / 256)), 256, 0, stream>>>(ST, stb, st_n);
    cvt_kernel <<<dim3((unsigned)(v_n / 256)),  256, 0, stream>>>(V, vb, v_n);

    scores_kernel<<<dim3(T_ / BN, T_ / BM, BH_), 128, 0, stream>>>(qr, sc);
    output_kernel<<<dim3(D_ / BN, T_ / BM, BH_), 128, 0, stream>>>(sc, vb, qr, stb, out);
    state_kernel <<<dim3(D_ / BN, N_ / BM, BH_), 128, 0, stream>>>(qr, vb, ST, out2);
}
